// BlockMoE_67465346286037
// MI455X (gfx1250) — compile-verified, matrix operand order checked by reading, not run on hardware
//
#include <hip/hip_runtime.h>

typedef __bf16 bf16;
typedef __attribute__((ext_vector_type(16))) __bf16 v16bf;
typedef __attribute__((ext_vector_type(8)))  float  v8f;
typedef __attribute__((ext_vector_type(4)))  int    v4i;

#define NTOK   8192
#define DIM    768
#define D3     2304
#define HEADS  12
#define HD     64
#define SEQ    1024
#define HID    3072
#define NEXP   8

#if __has_builtin(__builtin_amdgcn_global_load_async_to_lds_b128) && \
    __has_builtin(__builtin_amdgcn_s_wait_asynccnt)
#define HAS_ASYNC_LDS 1
#else
#define HAS_ASYNC_LDS 0
#endif

#define GAS __attribute__((address_space(1)))
#define LAS __attribute__((address_space(3)))

// ---------------------------------------------------------------- helpers
__device__ inline v8f wmma_bf16(v16bf a, v16bf b, v8f c) {
  return __builtin_amdgcn_wmma_f32_16x16x32_bf16(false, a, false, b, (short)0, c,
                                                 false, false);
}
__device__ inline v8f v8f_zero() { v8f z; for (int i = 0; i < 8; i++) z[i] = 0.f; return z; }

// LDS fragment load: tile stored [rows][ldk] bf16; lane l reads row r0+(l&15),
// K halves split across lane groups (contiguous k = 16h + s, same order for A and B).
__device__ inline v16bf ldsfrag(const bf16* base, int ldk, int r0, int k0) {
  int l = threadIdx.x & 31;
  const bf16* p = base + (size_t)(r0 + (l & 15)) * ldk + k0 + ((l >> 4) << 4);
  return *reinterpret_cast<const v16bf*>(p);
}
struct U8 { uint4 a, b; };
__device__ inline U8 ldg16(const bf16* src) {
  U8 r;
  const uint4* s = reinterpret_cast<const uint4*>(src);
  r.a = s[0]; r.b = s[1];
  return r;
}
__device__ inline void stl16(bf16* dst, U8 v) {
  uint4* d = reinterpret_cast<uint4*>(dst);
  d[0] = v.a; d[1] = v.b;
}
__device__ inline void copy16(bf16* dst, const bf16* src) { stl16(dst, ldg16(src)); }
__device__ inline void copy8(bf16* dst, const bf16* src) {
  *reinterpret_cast<uint4*>(dst) = *reinterpret_cast<const uint4*>(src);
}
__device__ inline void zero16(bf16* dst) {
  uint4 z = {0, 0, 0, 0};
  uint4* d = reinterpret_cast<uint4*>(dst);
  d[0] = z; d[1] = z;
}

#if HAS_ASYNC_LDS
// async DMA: global -> LDS, tracked by ASYNCcnt (no VGPR round-trip)
__device__ inline void async_cp8(bf16* lds_dst, const bf16* gsrc) {
  __builtin_amdgcn_global_load_async_to_lds_b128(
      (GAS v4i*)gsrc, (LAS v4i*)lds_dst, 0, 0);
}
__device__ inline void async_cp16(bf16* lds_dst, const bf16* gsrc) {
  async_cp8(lds_dst, gsrc);
  async_cp8(lds_dst + 8, gsrc + 8);
}
__device__ inline void async_wait() { __builtin_amdgcn_s_wait_asynccnt(0); }
#endif

// ---------------------------------------------------------------- transpose+cast
// in fp32 [Z][R][C] row-major -> out bf16 [Z][C][R]
__global__ void tcast_kernel(const float* __restrict__ in, bf16* __restrict__ out,
                             int R, int C) {
  size_t z = blockIdx.z;
  int i = blockIdx.x * 256 + threadIdx.x;
  if (i >= R * C) return;
  int r = i / C, c = i % C;
  out[z * (size_t)R * C + (size_t)c * R + r] = (bf16)in[z * (size_t)R * C + i];
}

// ---------------------------------------------------------------- layernorm -> bf16
__global__ void ln_kernel(const float* __restrict__ x, const float* __restrict__ g,
                          const float* __restrict__ b, bf16* __restrict__ out) {
  __shared__ float red[256];
  int row = blockIdx.x, tid = threadIdx.x;
  const float* xr = x + (size_t)row * DIM;
  float s = 0.f;
  for (int c = tid; c < DIM; c += 256) s += xr[c];
  red[tid] = s; __syncthreads();
  for (int st = 128; st > 0; st >>= 1) { if (tid < st) red[tid] += red[tid + st]; __syncthreads(); }
  float mean = red[0] / DIM; __syncthreads();
  float v = 0.f;
  for (int c = tid; c < DIM; c += 256) { float d = xr[c] - mean; v += d * d; }
  red[tid] = v; __syncthreads();
  for (int st = 128; st > 0; st >>= 1) { if (tid < st) red[tid] += red[tid + st]; __syncthreads(); }
  float rstd = rsqrtf(red[0] / DIM + 1e-5f);
  for (int c = tid; c < DIM; c += 256)
    out[(size_t)row * DIM + c] = (bf16)((xr[c] - mean) * rstd * g[c] + b[c]);
}

// ---------------------------------------------------------------- generic WMMA GEMM
// C[M,N] = act(A[M,K] @ Bt[N,K]^T + bias [+ resid])
// SEG: per-expert row segments via blockIdx.z; GATHER: A rows via index list.
// Double-buffered LDS; async global->LDS DMA when available; one barrier per K-step.
#define BM 128
#define BN 128
#define BKS 32
template <bool SEG, bool GATHER, bool BIAS, bool RESID, bool GELU, bool OUTF>
__global__ void gemm_kernel(const bf16* __restrict__ A, const bf16* __restrict__ Bt,
                            int M, int N, int K,
                            const int* __restrict__ segCnt, const int* __restrict__ segBase,
                            const int* __restrict__ gather, long bStrideZ,
                            const float* __restrict__ bias, int biasStrideZ,
                            const float* __restrict__ resid,
                            float* __restrict__ outF, bf16* __restrict__ outB) {
  __shared__ __attribute__((aligned(32))) bf16 As[2][BM * BKS];
  __shared__ __attribute__((aligned(32))) bf16 Bs[2][BN * BKS];
  int e = SEG ? blockIdx.z : 0;
  int mbase = blockIdx.x * BM, nbase = blockIdx.y * BN;
  int Meff = SEG ? segCnt[e] : M;
  if (SEG && mbase >= Meff) return;
  int segoff = SEG ? segBase[e] : 0;
  const bf16* Be = Bt + (size_t)e * bStrideZ;
  int tid = threadIdx.x, lane = tid & 31, w = tid >> 5;
  int wm = (w >> 1) * 32, wn = (w & 1) * 64;
  v8f acc[8];
  for (int i = 0; i < 8; i++) acc[i] = v8f_zero();

  int lr = tid >> 1, half = (tid & 1) * 16;
  int gm = mbase + lr;
  bool valid = SEG ? (gm < Meff) : true;
  long arow = 0;
  if (valid) {
    if (GATHER)   arow = gather[segoff + gm];
    else if (SEG) arow = segoff + gm;
    else          arow = gm;
  }
  const bf16* asrc = A + (size_t)arow * K + half;
  const bf16* bsrc = Be + (size_t)(nbase + lr) * K + half;
  bf16* adst = &As[0][0] + lr * BKS + half;
  bf16* bdst = &Bs[0][0] + lr * BKS + half;
  const int BUF = BM * BKS;
  const int nk = K / BKS;

  if (SEG && !valid) { zero16(adst); zero16(adst + BUF); }  // loop-invariant zero rows

#if HAS_ASYNC_LDS
  if (valid) async_cp16(adst, asrc);
  async_cp16(bdst, bsrc);
  async_wait();
  __syncthreads();
  for (int k = 0; k < nk; k++) {
    int cur = k & 1;
    bool more = (k + 1) < nk;
    if (more) {                       // DMA tile k+1 into the other buffer
      int nxt = cur ^ 1;
      if (valid) async_cp16(adst + nxt * BUF, asrc + (k + 1) * BKS);
      async_cp16(bdst + nxt * BUF, bsrc + (k + 1) * BKS);
    }
    const bf16* ab = &As[cur][0];
    const bf16* bb = &Bs[cur][0];
    v16bf af[2], bfr[4];
    for (int mi = 0; mi < 2; mi++) af[mi] = ldsfrag(ab, BKS, wm + mi * 16, 0);
    for (int ni = 0; ni < 4; ni++) bfr[ni] = ldsfrag(bb, BKS, wn + ni * 16, 0);
    for (int mi = 0; mi < 2; mi++)
      for (int ni = 0; ni < 4; ni++)
        acc[mi * 4 + ni] = wmma_bf16(af[mi], bfr[ni], acc[mi * 4 + ni]);
    if (more) { async_wait(); __syncthreads(); }
  }
#else
  if (valid) stl16(adst, ldg16(asrc));
  stl16(bdst, ldg16(bsrc));
  __syncthreads();
  for (int k = 0; k < nk; k++) {
    int cur = k & 1;
    bool more = (k + 1) < nk;
    U8 pa, pb;
    if (more) {                      // register-prefetch tile k+1 while WMMAs run
      if (valid) pa = ldg16(asrc + (k + 1) * BKS);
      pb = ldg16(bsrc + (k + 1) * BKS);
    }
    const bf16* ab = &As[cur][0];
    const bf16* bb = &Bs[cur][0];
    v16bf af[2], bfr[4];
    for (int mi = 0; mi < 2; mi++) af[mi] = ldsfrag(ab, BKS, wm + mi * 16, 0);
    for (int ni = 0; ni < 4; ni++) bfr[ni] = ldsfrag(bb, BKS, wn + ni * 16, 0);
    for (int mi = 0; mi < 2; mi++)
      for (int ni = 0; ni < 4; ni++)
        acc[mi * 4 + ni] = wmma_bf16(af[mi], bfr[ni], acc[mi * 4 + ni]);
    if (more) {
      int nxt = cur ^ 1;
      if (valid) stl16(adst + nxt * BUF, pa);
      stl16(bdst + nxt * BUF, pb);
      __syncthreads();
    }
  }
#endif

  // specialized epilogue (compile-time flags -> no accumulator spills)
  for (int mi = 0; mi < 2; mi++)
    for (int ni = 0; ni < 4; ni++) {
      v8f c = acc[mi * 4 + ni];
      int colb = nbase + wn + ni * 16 + (lane & 15);
      float bv = BIAS ? bias[(size_t)e * biasStrideZ + colb] : 0.f;
      for (int r = 0; r < 8; r++) {
        int lm = wm + mi * 16 + r + ((lane >> 4) << 3);
        int gmr = mbase + lm;
        if (SEG && gmr >= Meff) continue;
        long orow = SEG ? (long)(segoff + gmr) : (long)gmr;
        float v = c[r] + bv;
        if (RESID) v += resid[(size_t)orow * N + colb];
        if (GELU) v = 0.5f * v * (1.f + erff(v * 0.70710678f));
        if (OUTF) outF[(size_t)orow * N + colb] = v;
        else      outB[(size_t)orow * N + colb] = (bf16)v;
      }
    }
}

// ---------------------------------------------------------------- flash attention
// grid (B*HEADS, SEQ/64); per-block: 64 q rows, stream 32 kv rows per iter.
__global__ void attn_kernel(const bf16* __restrict__ qkv, bf16* __restrict__ out) {
  __shared__ __attribute__((aligned(32))) bf16 Qs[64 * 64];
  __shared__ __attribute__((aligned(32))) bf16 Ks[32 * 64];
  __shared__ __attribute__((aligned(32))) bf16 Vts[64 * 32];
  __shared__ __attribute__((aligned(32))) bf16 Ps[64 * 32];
  __shared__ float Ss[64 * 32];
  __shared__ float mrow[64], lrow[64], arow[64];

  int bh = blockIdx.x;
  int b = bh / HEADS, h = bh % HEADS;
  int q0 = blockIdx.y * 64;
  int tid = threadIdx.x, lane = tid & 31, w = tid >> 5;
  size_t tokBase = (size_t)b * SEQ;

  { int r = tid >> 2, seg = (tid & 3) * 16;
    const bf16* src = qkv + (tokBase + q0 + r) * D3 + h * HD + seg;
#if HAS_ASYNC_LDS
    async_cp16(Qs + r * 64 + seg, src);
#else
    copy16(Qs + r * 64 + seg, src);
#endif
  }
  if (tid < 64) { mrow[tid] = -1e30f; lrow[tid] = 0.f; }

  v8f o0 = v8f_zero(), o1 = v8f_zero();
  int qm = (w & 3) * 16;
  int kn = (w >> 2) * 16;
  int vn0 = (w >> 2) * 32;
#if HAS_ASYNC_LDS
  async_wait();
#endif
  __syncthreads();

  for (int kv = 0; kv < SEQ; kv += 32) {
    { int r = tid >> 3, seg = (tid & 7) * 8;
      const bf16* src = qkv + (tokBase + kv + r) * D3 + DIM + h * HD + seg;
#if HAS_ASYNC_LDS
      async_cp8(Ks + r * 64 + seg, src);
#else
      copy8(Ks + r * 64 + seg, src);
#endif
    }
    { int d = tid >> 2, rg = (tid & 3) * 8;
      for (int i = 0; i < 8; i++) {
        int r = rg + i;
        Vts[d * 32 + r] = qkv[(tokBase + kv + r) * D3 + 2 * DIM + h * HD + d];
      } }
#if HAS_ASYNC_LDS
    async_wait();
#endif
    __syncthreads();

    v8f s = v8f_zero();
    for (int ks = 0; ks < 2; ks++) {
      v16bf a  = ldsfrag(Qs, 64, qm, ks * 32);
      v16bf bb = ldsfrag(Ks, 64, kn, ks * 32);
      s = wmma_bf16(a, bb, s);
    }
    for (int r = 0; r < 8; r++) {
      int row = qm + r + ((lane >> 4) << 3);
      int col = kn + (lane & 15);
      Ss[row * 32 + col] = s[r] * 0.125f;   // hd^-0.5
    }
    __syncthreads();

    if (tid < 64) {
      float mo = mrow[tid], mx = mo;
      for (int c = 0; c < 32; c++) mx = fmaxf(mx, Ss[tid * 32 + c]);
      float al = expf(mo - mx), sum = 0.f;
      for (int c = 0; c < 32; c++) {
        float p = expf(Ss[tid * 32 + c] - mx);
        sum += p;
        Ps[tid * 32 + c] = (bf16)p;
      }
      mrow[tid] = mx; lrow[tid] = lrow[tid] * al + sum; arow[tid] = al;
    }
    __syncthreads();

    { v16bf a  = ldsfrag(Ps, 32, qm, 0);
      v16bf b0 = ldsfrag(Vts, 32, vn0, 0);
      v16bf b1 = ldsfrag(Vts, 32, vn0 + 16, 0);
      for (int r = 0; r < 8; r++) {
        float al = arow[qm + r + ((lane >> 4) << 3)];
        o0[r] *= al; o1[r] *= al;
      }
      o0 = wmma_bf16(a, b0, o0);
      o1 = wmma_bf16(a, b1, o1);
    }
    __syncthreads();
  }
  for (int r = 0; r < 8; r++) {
    int row = qm + r + ((lane >> 4) << 3);
    float inv = 1.f / lrow[row];
    int col0 = vn0 + (lane & 15);
    size_t base = (tokBase + q0 + row) * DIM + h * HD;
    out[base + col0]      = (bf16)(o0[r] * inv);
    out[base + col0 + 16] = (bf16)(o1[r] * inv);
  }
}

// ---------------------------------------------------------------- MoE routing
__global__ void route_kernel(const bf16* __restrict__ h2, const float* __restrict__ gw,
                             const float* __restrict__ gb, int* __restrict__ cnt,
                             int* __restrict__ selA, int* __restrict__ idxA,
                             float* __restrict__ wtA) {
  int n = blockIdx.x * blockDim.x + threadIdx.x;
  if (n >= NTOK) return;
  float lg[NEXP];
  for (int e = 0; e < NEXP; e++) lg[e] = gb[e];
  const bf16* hr = h2 + (size_t)n * DIM;
  for (int c = 0; c < DIM; c++) {
    float hv = (float)hr[c];
    const float* gr = gw + c * NEXP;
    for (int e = 0; e < NEXP; e++) lg[e] += hv * gr[e];
  }
  int e0 = 0;
  for (int e = 1; e < NEXP; e++) if (lg[e] > lg[e0]) e0 = e;     // first max (top_k tie rule)
  int e1 = (e0 == 0) ? 1 : 0;
  for (int e = 0; e < NEXP; e++) if (e != e0 && lg[e] > lg[e1]) e1 = e;
  float z1 = expf(lg[e1] - lg[e0]);
  float inv = 1.f / (1.f + z1);
  int i0 = atomicAdd(&cnt[e0], 1);
  int i1 = atomicAdd(&cnt[e1], 1);
  selA[n * 2] = e0; selA[n * 2 + 1] = e1;
  idxA[n * 2] = i0; idxA[n * 2 + 1] = i1;
  wtA[n * 2] = inv; wtA[n * 2 + 1] = z1 * inv;
}

__global__ void scan_kernel(const int* __restrict__ cnt, int* __restrict__ base) {
  if (threadIdx.x == 0) {
    int a = 0;
    for (int e = 0; e < NEXP; e++) { base[e] = a; a += cnt[e]; }
  }
}

__global__ void slot_kernel(const int* __restrict__ selA, const int* __restrict__ idxA,
                            const int* __restrict__ base, int* __restrict__ rowidx,
                            int* __restrict__ tok2slot) {
  int i = blockIdx.x * blockDim.x + threadIdx.x;
  if (i >= NTOK * 2) return;
  int slot = base[selA[i]] + idxA[i];
  rowidx[slot] = i >> 1;
  tok2slot[i] = slot;
}

__global__ void combine_kernel(const float* __restrict__ x2, const float* __restrict__ y,
                               const int* __restrict__ tok2slot, const float* __restrict__ wtA,
                               float* __restrict__ out) {
  int n = blockIdx.x, tid = threadIdx.x;
  int s0 = tok2slot[n * 2], s1 = tok2slot[n * 2 + 1];
  float w0 = wtA[n * 2], w1 = wtA[n * 2 + 1];
  for (int c = tid; c < DIM; c += 256)
    out[(size_t)n * DIM + c] = x2[(size_t)n * DIM + c]
                             + w0 * y[(size_t)s0 * DIM + c]
                             + w1 * y[(size_t)s1 * DIM + c];
}

// ---------------------------------------------------------------- host
extern "C" void kernel_launch(void* const* d_in, const int* in_sizes, int n_in,
                              void* d_out, int out_size, void* d_ws, size_t ws_size,
                              hipStream_t stream) {
  const float* x      = (const float*)d_in[0];
  const float* ln1_g  = (const float*)d_in[1];
  const float* ln1_b  = (const float*)d_in[2];
  const float* qkv_w  = (const float*)d_in[3];
  const float* proj_w = (const float*)d_in[4];
  const float* proj_b = (const float*)d_in[5];
  const float* ln2_g  = (const float*)d_in[6];
  const float* ln2_b  = (const float*)d_in[7];
  const float* gate_w = (const float*)d_in[8];
  const float* gate_b = (const float*)d_in[9];
  const float* w1     = (const float*)d_in[10];
  const float* b1     = (const float*)d_in[11];
  const float* w2     = (const float*)d_in[12];
  const float* b2     = (const float*)d_in[13];
  float* out = (float*)d_out;

  char* p = (char*)d_ws;
  auto alloc = [&](size_t bytes) {
    char* r = p;
    p += (bytes + 255) & ~(size_t)255;
    return r;
  };
  bf16*  qkvT   = (bf16*)alloc((size_t)D3 * DIM * 2);
  bf16*  projT  = (bf16*)alloc((size_t)DIM * DIM * 2);
  bf16*  w1T    = (bf16*)alloc((size_t)NEXP * DIM * HID * 2);
  bf16*  w2T    = (bf16*)alloc((size_t)NEXP * HID * DIM * 2);
  bf16*  hb     = (bf16*)alloc((size_t)NTOK * DIM * 2);
  bf16*  qkvb   = (bf16*)alloc((size_t)NTOK * D3 * 2);
  bf16*  attnb  = (bf16*)alloc((size_t)NTOK * DIM * 2);
  float* x2     = (float*)alloc((size_t)NTOK * DIM * 4);
  bf16*  h2b    = (bf16*)alloc((size_t)NTOK * DIM * 2);
  bf16*  he     = (bf16*)alloc((size_t)NTOK * 2 * HID * 2);
  float* yv     = (float*)alloc((size_t)NTOK * 2 * DIM * 4);
  int*   cnt    = (int*)alloc(NEXP * 4);
  int*   segb   = (int*)alloc(NEXP * 4);
  int*   selA   = (int*)alloc((size_t)NTOK * 2 * 4);
  int*   idxA   = (int*)alloc((size_t)NTOK * 2 * 4);
  float* wtA    = (float*)alloc((size_t)NTOK * 2 * 4);
  int*   rowidx = (int*)alloc((size_t)NTOK * 2 * 4);
  int*   t2s    = (int*)alloc((size_t)NTOK * 2 * 4);
  (void)ws_size; (void)n_in; (void)in_sizes; (void)out_size;

  (void)hipMemsetAsync(cnt, 0, NEXP * sizeof(int), stream);

  // weight transpose+cast to bf16 (N-major for WMMA B fragments)
  tcast_kernel<<<dim3((DIM * D3 + 255) / 256, 1, 1), 256, 0, stream>>>(qkv_w, qkvT, DIM, D3);
  tcast_kernel<<<dim3((DIM * DIM + 255) / 256, 1, 1), 256, 0, stream>>>(proj_w, projT, DIM, DIM);
  tcast_kernel<<<dim3((DIM * HID + 255) / 256, 1, NEXP), 256, 0, stream>>>(w1, w1T, DIM, HID);
  tcast_kernel<<<dim3((HID * DIM + 255) / 256, 1, NEXP), 256, 0, stream>>>(w2, w2T, HID, DIM);

  // LN1 -> bf16
  ln_kernel<<<NTOK, 256, 0, stream>>>(x, ln1_g, ln1_b, hb);

  // QKV GEMM -> bf16 [N, 2304]
  gemm_kernel<false, false, false, false, false, false>
      <<<dim3(NTOK / BM, D3 / BN, 1), 256, 0, stream>>>(
      hb, qkvT, NTOK, D3, DIM, nullptr, nullptr, nullptr, 0,
      nullptr, 0, nullptr, nullptr, qkvb);

  // flash attention -> bf16 [N, 768]
  attn_kernel<<<dim3(8 * HEADS, SEQ / 64, 1), 256, 0, stream>>>(qkvb, attnb);

  // proj + bias + residual -> x2 fp32
  gemm_kernel<false, false, true, true, false, true>
      <<<dim3(NTOK / BM, DIM / BN, 1), 256, 0, stream>>>(
      attnb, projT, NTOK, DIM, DIM, nullptr, nullptr, nullptr, 0,
      proj_b, 0, x, x2, nullptr);

  // LN2 -> bf16
  ln_kernel<<<NTOK, 256, 0, stream>>>(x2, ln2_g, ln2_b, h2b);

  // routing
  route_kernel<<<NTOK / 128, 128, 0, stream>>>(h2b, gate_w, gate_b, cnt, selA, idxA, wtA);
  scan_kernel<<<1, 32, 0, stream>>>(cnt, segb);
  slot_kernel<<<(NTOK * 2) / 256, 256, 0, stream>>>(selA, idxA, segb, rowidx, t2s);

  // expert GEMM1: gather(h2) @ w1[e] + b1 -> gelu -> he bf16
  gemm_kernel<true, true, true, false, true, false>
      <<<dim3(NTOK / BM, HID / BN, NEXP), 256, 0, stream>>>(
      h2b, w1T, NTOK, HID, DIM, cnt, segb, rowidx, (long)HID * DIM,
      b1, HID, nullptr, nullptr, he);

  // expert GEMM2: he @ w2[e] + b2 -> y fp32
  gemm_kernel<true, false, true, false, false, true>
      <<<dim3(NTOK / BM, DIM / BN, NEXP), 256, 0, stream>>>(
      he, w2T, NTOK, DIM, HID, cnt, segb, nullptr, (long)DIM * HID,
      b2, DIM, nullptr, yv, nullptr);

  // out = x2 + w0*y[slot0] + w1*y[slot1]
  combine_kernel<<<NTOK, 256, 0, stream>>>(x2, yv, t2s, wtA, out);
}
